// MaskedSelfAttention_1125281431641
// MI455X (gfx1250) — matrix operand order repaired and checked by reading, not compiled, not run on hardware
//
#include <hip/hip_runtime.h>
#include <hip/hip_bf16.h>

typedef __attribute__((ext_vector_type(16))) _Float16 v16h;
typedef __attribute__((ext_vector_type(8)))  _Float16 v8h;
typedef __attribute__((ext_vector_type(8)))  float    v8f;

#define CDIM   64
#define NSEQ   4096
#define NBATCH 8

static __device__ __forceinline__ v16h hcat(v8h lo, v8h hi) {
  return __builtin_shufflevector(lo, hi, 0,1,2,3,4,5,6,7,8,9,10,11,12,13,14,15);
}

static __device__ __forceinline__ v8f zero8() {
  v8f z = {0.f,0.f,0.f,0.f,0.f,0.f,0.f,0.f};
  return z;
}

// A fragment (16x32 f16, MxK) from row-major f16 src (leading dim ld).
// Lanes 0-15: row, K = {kbase+0..7, kbase+16..23}; lanes 16-31: K = {+8..15, +24..31}.
static __device__ __forceinline__ v16h load_a16(const _Float16* src, int row, int ld,
                                                int kbase, int lane) {
  const int half = lane >> 4;
  const _Float16* p = src + (size_t)row * ld + kbase + half * 8;
  v8h lo = *(const v8h*)(p);
  v8h hi = *(const v8h*)(p + 16);
  return hcat(lo, hi);
}

// B fragment (32x16 f16, KxN): element (k, c) at src[(cbase+c)*ld + kbase + k].
// Lanes 0-15 hold K = {0..7, 16..23} of their column; lanes 16-31 hold K = {8..15, 24..31}.
static __device__ __forceinline__ v16h load_b16(const _Float16* src, int cbase, int ld,
                                                int kbase, int lane) {
  const int half = lane >> 4;
  const int col  = lane & 15;
  const _Float16* p = src + (size_t)(cbase + col) * ld + kbase + half * 8;
  v8h lo = *(const v8h*)(p);
  v8h hi = *(const v8h*)(p + 16);
  return hcat(lo, hi);
}

static __device__ __forceinline__ v8f wmma16(v16h a, v16h b, v8f c) {
  return __builtin_amdgcn_wmma_f32_16x16x32_f16(false, a, false, b, (short)0, c,
                                                false, false);
}

// ---------------------------------------------------------------------------
// Kernel 0a: x [B,C,N] f32  ->  xT [B,N,C] f16   (LDS-tiled transpose+convert)
// ---------------------------------------------------------------------------
__global__ void __launch_bounds__(256) xpose_cvt(const float* __restrict__ x,
                                                 _Float16* __restrict__ xt) {
  __shared__ float tile[64][65];
  const int b   = blockIdx.y;
  const int nb  = blockIdx.x * 64;
  const int tid = threadIdx.x;
  const int cc  = tid & 63;
  const int rr  = tid >> 6;   // 0..3
  const float* xb = x + (size_t)b * CDIM * NSEQ;
  #pragma unroll
  for (int i = 0; i < 16; ++i) {
    int c = i * 4 + rr;
    tile[c][cc] = xb[(size_t)c * NSEQ + nb + cc];
  }
  __syncthreads();
  _Float16* xtb = xt + (size_t)b * NSEQ * CDIM;
  #pragma unroll
  for (int i = 0; i < 16; ++i) {
    int n = i * 4 + rr;
    xtb[(size_t)(nb + n) * CDIM + cc] = (_Float16)tile[cc][n];
  }
}

// ---------------------------------------------------------------------------
// Kernel 0b: one-time f32 -> f16 conversion of weight matrices.
// ---------------------------------------------------------------------------
__global__ void __launch_bounds__(256) cvt_weights(const float* __restrict__ w_qkv,
                                                   const float* __restrict__ w_proj,
                                                   _Float16* __restrict__ wq16,
                                                   _Float16* __restrict__ wp16) {
  const int i = blockIdx.x * 256 + threadIdx.x;
  if (i < 192 * 64) wq16[i] = (_Float16)w_qkv[i];
  if (i < 64 * 64)  wp16[i] = (_Float16)w_proj[i];
}

// ---------------------------------------------------------------------------
// Kernel 1: QKV projection.  One wave = 16 rows of xT, 12 output col-tiles.
//   Q (pre-scaled by 1/sqrt(C)), K stored [B,N,C] f16.  V stored [B,C,N] f16.
// ---------------------------------------------------------------------------
__global__ void __launch_bounds__(256) qkv_proj(const _Float16* __restrict__ xt,
                                                const _Float16* __restrict__ wq16,
                                                _Float16* __restrict__ qo,
                                                _Float16* __restrict__ ko,
                                                _Float16* __restrict__ vto) {
  const int lane = threadIdx.x & 31;
  const int wave = threadIdx.x >> 5;
  const int tileId = blockIdx.x * 8 + wave;   // over B*N/16 = 2048 tiles
  const int b  = tileId >> 8;                 // 256 tiles per batch
  const int nb = (tileId & 255) << 4;
  const int half = lane >> 4;
  const int l15  = lane & 15;

  const _Float16* xb = xt + (size_t)b * NSEQ * CDIM;
  v16h a0 = load_a16(xb, nb + l15, CDIM, 0,  lane);
  v16h a1 = load_a16(xb, nb + l15, CDIM, 32, lane);

  #pragma unroll
  for (int jt = 0; jt < 12; ++jt) {
    v8f acc = zero8();
    acc = wmma16(a0, load_b16(wq16, jt * 16, CDIM, 0,  lane), acc);
    acc = wmma16(a1, load_b16(wq16, jt * 16, CDIM, 32, lane), acc);
    const int j = jt * 16 + l15;      // 0..191
    if (jt < 4) {                     // Q rows 0..63: fold in softmax scale
      _Float16* dst = qo + (size_t)b * NSEQ * CDIM;
      #pragma unroll
      for (int r = 0; r < 8; ++r)
        dst[(size_t)(nb + r + half * 8) * CDIM + j] = (_Float16)(acc[r] * 0.125f);
    } else if (jt < 8) {              // K rows 64..127
      _Float16* dst = ko + (size_t)b * NSEQ * CDIM;
      const int cj = j - 64;
      #pragma unroll
      for (int r = 0; r < 8; ++r)
        dst[(size_t)(nb + r + half * 8) * CDIM + cj] = (_Float16)acc[r];
    } else {                          // V rows 128..191, stored transposed
      const int cj = j - 128;
      v8h pk;
      #pragma unroll
      for (int r = 0; r < 8; ++r) pk[r] = (_Float16)acc[r];
      *(v8h*)(vto + (size_t)b * CDIM * NSEQ + (size_t)cj * NSEQ + nb + half * 8) = pk;
    }
  }
}

// ---------------------------------------------------------------------------
// Kernel 2: fused masked flash-attention + output projection.
//   One wave = 16 queries.  Key loop in tiles of 128 (8 S tiles, 32 WMMA/step).
//   Deep-preloaded B fragments so loads run 8-deep ahead of their WMMA.
// ---------------------------------------------------------------------------
#define KT 128

__global__ void __launch_bounds__(256) attn_fused(const _Float16* __restrict__ qm,
                                                  const _Float16* __restrict__ km,
                                                  const _Float16* __restrict__ vtm,
                                                  const int* __restrict__ fg,
                                                  const _Float16* __restrict__ wp16,
                                                  const float* __restrict__ b_proj,
                                                  float* __restrict__ out) {
  __shared__ _Float16 ldsP[8][16 * KT];     // 4KB per wave = 32KB
  const int lane = threadIdx.x & 31;
  const int wave = threadIdx.x >> 5;
  const int b    = blockIdx.y;
  const int qb   = blockIdx.x * 128 + wave * 16;
  const int half = lane >> 4;
  const int l15  = lane & 15;

  const _Float16* Q   = qm  + (size_t)b * NSEQ * CDIM;
  const _Float16* K   = km  + (size_t)b * NSEQ * CDIM;
  const _Float16* V   = vtm + (size_t)b * CDIM * NSEQ;  // [C][N]
  const int*      fgb = fg  + (size_t)b * NSEQ;
  _Float16* P = &ldsP[wave][0];

  v16h aq0 = load_a16(Q, qb + l15, CDIM, 0,  lane);
  v16h aq1 = load_a16(Q, qb + l15, CDIM, 32, lane);

  v8f o[4];
  float mrow[8], lrow[8];
  #pragma unroll
  for (int t = 0; t < 4; ++t) o[t] = zero8();
  #pragma unroll
  for (int r = 0; r < 8; ++r) { mrow[r] = -1e30f; lrow[r] = 0.f; }

  for (int kb = 0; kb < NSEQ; kb += KT) {
    // hint next K/V tiles toward the caches
    if (kb + KT < NSEQ) {
      __builtin_prefetch((const char*)(K + (size_t)(kb + KT) * CDIM) + lane * 512, 0, 3);
      __builtin_prefetch((const char*)(V + (size_t)(lane * 2) * NSEQ + kb + KT), 0, 3);
    }
    const _Float16* Kb = K + (size_t)kb * CDIM;

    // --- S = Q K^T for 8 key tiles of 16 (Q already carries 1/sqrt(C)) ---
    // Preload all k=0..31 fragments, consume them while streaming in k=32..63.
    v16h bk[8];
    #pragma unroll
    for (int t = 0; t < 8; ++t) bk[t] = load_b16(Kb, t * 16, CDIM, 0, lane);
    v8f s[8];
    #pragma unroll
    for (int t = 0; t < 8; ++t) {
      s[t] = wmma16(aq0, bk[t], zero8());
      bk[t] = load_b16(Kb, t * 16, CDIM, 32, lane);   // refill after use
    }
    #pragma unroll
    for (int t = 0; t < 8; ++t)
      s[t] = wmma16(aq1, bk[t], s[t]);

    // key mask (column key index == l15 in D layout)
    #pragma unroll
    for (int t = 0; t < 8; ++t) {
      const int keep = fgb[kb + t * 16 + l15];
      #pragma unroll
      for (int r = 0; r < 8; ++r)
        s[t][r] = keep ? s[t][r] : -1e30f;
    }
    // row max (rows live in a 16-lane half -> xor 1,2,4,8)
    float mnew[8];
    #pragma unroll
    for (int r = 0; r < 8; ++r) {
      float a01 = fmaxf(s[0][r], s[1][r]), a23 = fmaxf(s[2][r], s[3][r]);
      float a45 = fmaxf(s[4][r], s[5][r]), a67 = fmaxf(s[6][r], s[7][r]);
      mnew[r] = fmaxf(fmaxf(a01, a23), fmaxf(a45, a67));
    }
    #pragma unroll
    for (int xm = 1; xm <= 8; xm <<= 1)
      #pragma unroll
      for (int r = 0; r < 8; ++r)
        mnew[r] = fmaxf(mnew[r], __shfl_xor(mnew[r], xm, 32));
    #pragma unroll
    for (int r = 0; r < 8; ++r) mnew[r] = fmaxf(mnew[r], mrow[r]);
    // exponentiate + partial row sums
    float lad[8];
    #pragma unroll
    for (int r = 0; r < 8; ++r) lad[r] = 0.f;
    #pragma unroll
    for (int t = 0; t < 8; ++t)
      #pragma unroll
      for (int r = 0; r < 8; ++r) {
        float p = __expf(s[t][r] - mnew[r]);
        s[t][r] = p;
        lad[r] += p;
      }
    #pragma unroll
    for (int xm = 1; xm <= 8; xm <<= 1)
      #pragma unroll
      for (int r = 0; r < 8; ++r)
        lad[r] += __shfl_xor(lad[r], xm, 32);
    // online-softmax rescale of running state
    #pragma unroll
    for (int r = 0; r < 8; ++r) {
      float sc = __expf(mrow[r] - mnew[r]);
      lrow[r] = lrow[r] * sc + lad[r];
      mrow[r] = mnew[r];
      #pragma unroll
      for (int t = 0; t < 4; ++t) o[t][r] *= sc;
    }
    // P: D layout -> LDS (row-major 16xKT f16) -> A fragments
    #pragma unroll
    for (int t = 0; t < 8; ++t)
      #pragma unroll
      for (int r = 0; r < 8; ++r)
        P[(r + half * 8) * KT + t * 16 + l15] = (_Float16)s[t][r];
    asm volatile("s_wait_dscnt 0x0" ::: "memory");
    v16h ap[4];
    #pragma unroll
    for (int ks = 0; ks < 4; ++ks)
      ap[ks] = load_a16(P, l15, KT, ks * 32, lane);
    // --- O += P V : per k-step preload 4 V fragments, 4 independent chains ---
    #pragma unroll
    for (int ks = 0; ks < 4; ++ks) {
      v16h bv0 = load_b16(V, 0,  NSEQ, kb + ks * 32, lane);
      v16h bv1 = load_b16(V, 16, NSEQ, kb + ks * 32, lane);
      v16h bv2 = load_b16(V, 32, NSEQ, kb + ks * 32, lane);
      v16h bv3 = load_b16(V, 48, NSEQ, kb + ks * 32, lane);
      o[0] = wmma16(ap[ks], bv0, o[0]);
      o[1] = wmma16(ap[ks], bv1, o[1]);
      o[2] = wmma16(ap[ks], bv2, o[2]);
      o[3] = wmma16(ap[ks], bv3, o[3]);
    }
    asm volatile("" ::: "memory");   // keep next-iter LDS stores after these loads
  }

  // normalize
  #pragma unroll
  for (int r = 0; r < 8; ++r) {
    float inv = 1.0f / lrow[r];
    #pragma unroll
    for (int t = 0; t < 4; ++t) o[t][r] *= inv;
  }
  // O -> LDS -> A fragments for the output projection
  #pragma unroll
  for (int ct = 0; ct < 4; ++ct)
    #pragma unroll
    for (int r = 0; r < 8; ++r)
      P[(r + half * 8) * CDIM + ct * 16 + l15] = (_Float16)o[ct][r];
  asm volatile("s_wait_dscnt 0x0" ::: "memory");
  v16h ao0 = load_a16(P, l15, CDIM, 0,  lane);
  v16h ao1 = load_a16(P, l15, CDIM, 32, lane);

  float* ob = out + (size_t)b * CDIM * NSEQ;   // [C][N]
  #pragma unroll
  for (int jt = 0; jt < 4; ++jt) {
    v8f acc = zero8();
    acc = wmma16(ao0, load_b16(wp16, jt * 16, CDIM, 0,  lane), acc);
    acc = wmma16(ao1, load_b16(wp16, jt * 16, CDIM, 32, lane), acc);
    const int j = jt * 16 + l15;
    const float bias = b_proj[j];
    float* dst = ob + (size_t)j * NSEQ + qb + half * 8;  // rows contiguous in n
    float4 v0 = make_float4(acc[0] + bias, acc[1] + bias, acc[2] + bias, acc[3] + bias);
    float4 v1 = make_float4(acc[4] + bias, acc[5] + bias, acc[6] + bias, acc[7] + bias);
    *(float4*)(dst)     = v0;
    *(float4*)(dst + 4) = v1;
  }
}

// ---------------------------------------------------------------------------
extern "C" void kernel_launch(void* const* d_in, const int* in_sizes, int n_in,
                              void* d_out, int out_size, void* d_ws, size_t ws_size,
                              hipStream_t stream) {
  const float* x      = (const float*)d_in[0];
  const int*   fg     = (const int*)  d_in[1];
  const float* w_qkv  = (const float*)d_in[2];
  const float* w_proj = (const float*)d_in[3];
  const float* b_proj = (const float*)d_in[4];
  float* out = (float*)d_out;

  const size_t elems = (size_t)NBATCH * NSEQ * CDIM;   // 2M f16 each = 4MB
  _Float16* xt   = (_Float16*)d_ws;
  _Float16* q    = xt + elems;
  _Float16* k    = q  + elems;
  _Float16* vt   = k  + elems;
  _Float16* wq16 = vt + elems;                 // 192*64 f16
  _Float16* wp16 = wq16 + (size_t)192 * 64;    // 64*64 f16

  dim3 blk(256);
  cvt_weights<<<dim3(48),                blk, 0, stream>>>(w_qkv, w_proj, wq16, wp16);
  xpose_cvt  <<<dim3(NSEQ / 64, NBATCH), blk, 0, stream>>>(x, xt);
  qkv_proj   <<<dim3((NBATCH * NSEQ) / 128), blk, 0, stream>>>(xt, wq16, q, k, vt);
  attn_fused <<<dim3(NSEQ / 128, NBATCH),    blk, 0, stream>>>(q, k, vt, fg,
                                                               wp16, b_proj, out);
}
